// SequentialVAE_68779606278781
// MI455X (gfx1250) — matrix operand order repaired and checked by reading, not compiled, not run on hardware
//
#include <hip/hip_runtime.h>
#include <hip/hip_bf16.h>

typedef __attribute__((ext_vector_type(16))) __bf16 v16bf;
typedef __attribute__((ext_vector_type(8)))  __bf16 v8bf;
typedef __attribute__((ext_vector_type(8)))  float  v8f;
typedef __attribute__((ext_vector_type(4)))  unsigned int u32x4;
typedef __attribute__((ext_vector_type(4)))  int i32x4;
typedef __attribute__((ext_vector_type(8)))  int i32x8;

#define T_LEN 8192
#define DH    512
#define DZ    32

// ---------------------------------------------------------------------------
// WMMA helper: D = A(16x32 bf16) * B(32x16 bf16) + C(16x16 f32)
// ---------------------------------------------------------------------------
__device__ __forceinline__ v8f wmma_bf16f32(v16bf a, v16bf b, v8f c) {
  return __builtin_amdgcn_wmma_f32_16x16x32_bf16(
      /*neg_a=*/false, a, /*neg_b=*/false, b,
      /*c_mod=*/(short)0, c, /*reuse_a=*/false, /*reuse_b=*/false);
}

// A-fragment (16x32, bf16, row-major source, ld = row stride in elems).
__device__ __forceinline__ v16bf load_a_frag(const __bf16* __restrict__ base,
                                             int ld, int m, int kb, int lane) {
  const int half = lane >> 4;
  const __bf16* p = base + m * ld + kb + half * 8;
  v8bf lo = *(const v8bf*)p;
  v8bf hi = *(const v8bf*)(p + 16);
  v16bf r;
#pragma unroll
  for (int i = 0; i < 8; i++) { r[i] = lo[i]; r[i + 8] = hi[i]; }
  return r;
}

// B-fragment (32x16) from N-major storage Bcol[n*ld + k].
__device__ __forceinline__ v16bf load_b_frag(const __bf16* __restrict__ base,
                                             int ld, int n, int kb, int lane) {
  const int half = lane >> 4;
  const __bf16* p = base + n * ld + kb + half * 8;
  v8bf lo = *(const v8bf*)p;
  v8bf hi = *(const v8bf*)(p + 16);
  v16bf r;
#pragma unroll
  for (int i = 0; i < 8; i++) { r[i] = lo[i]; r[i + 8] = hi[i]; }
  return r;
}

// ---------------------------------------------------------------------------
// TDM: async-load one contiguous 32x32 f32 block (4KB) into LDS.
// Descriptor per CDNA5 ISA:
//  group0: count=1, lds_addr, global_addr[56:0], type=2
//  group1: data_size=4B, tensor_dim0=1024, tensor_dim1=1, tile=1024x1,
//          stride0=1024, no multicast / barrier / pad / iterate.
// Tracked on TENSORcnt (in-order per wave). 6-arg builtin (clang-23 form).
// ---------------------------------------------------------------------------
__device__ __forceinline__ void tdm_load_block(const float* __restrict__ g,
                                               unsigned lds_off) {
  const unsigned long long ga = (unsigned long long)(uintptr_t)g;
  u32x4 g0;
  g0[0] = 1u;                                        // count=1, user mode
  g0[1] = lds_off;                                   // LDS byte address
  g0[2] = (unsigned)ga;                              // global_addr[31:0]
  g0[3] = (unsigned)((ga >> 32) & 0x1FFFFFFull)      // global_addr[56:32]
          | (2u << 30);                              // type=2 ("image")
  i32x8 g1;
  g1[0] = (int)(2u << 16);        // workgroup_mask=0, data_size=2 (4 bytes)
  g1[1] = (int)(1024u << 16);     // tensor_dim0[15:0]=1024 in bits[31:16]
  g1[2] = (int)(1u << 16);        // tensor_dim0[31:16]=0 | tensor_dim1[15:0]=1
  g1[3] = (int)(1024u << 16);     // tensor_dim1[31:16]=0 | tile_dim0=1024
  g1[4] = 1;                      // tile_dim1=1, tile_dim2=0
  g1[5] = 1024;                   // tensor_dim0_stride[31:0]
  g1[6] = 0;                      // stride0 hi | stride1 lo
  g1[7] = 0;                      // stride1 hi
  i32x4 z4 = {0, 0, 0, 0};
  i32x8 z8 = {0, 0, 0, 0, 0, 0, 0, 0};
  __builtin_amdgcn_tensor_load_to_lds(g0, g1, z4, z4, z8, 0);
}

// ---------------------------------------------------------------------------
// Precision-conversion kernels
// ---------------------------------------------------------------------------
__global__ __launch_bounds__(256) void cvt_bf16_kernel(
    const float* __restrict__ src, __bf16* __restrict__ dst, int n) {
  int i = blockIdx.x * blockDim.x + threadIdx.x;
  if (i < n) dst[i] = (__bf16)src[i];
}

__global__ __launch_bounds__(256) void cvt_transpose_kernel(
    const float* __restrict__ src, __bf16* __restrict__ dst, int K, int N) {
  int i = blockIdx.x * blockDim.x + threadIdx.x;
  if (i < K * N) {
    int k = i / N, n = i % N;
    dst[n * K + k] = (__bf16)src[i];
  }
}

// ---------------------------------------------------------------------------
// Generic WMMA GEMM: out = act(Abf(MxK) @ B + bias), B given as Bcol (NxK).
// One wave -> 16 x (16*NT) tile; operands L2-resident (192MB L2), no LDS.
// ---------------------------------------------------------------------------
template <int NT, bool RELU, bool OUTBF>
__global__ __launch_bounds__(128) void wmma_gemm_kernel(
    const __bf16* __restrict__ A, const __bf16* __restrict__ Bcol,
    const float* __restrict__ bias, float* __restrict__ outf,
    __bf16* __restrict__ outb, int M, int N, int K) {
  const int lane = threadIdx.x & 31;
  const int wid  = blockIdx.x * (blockDim.x >> 5) + (threadIdx.x >> 5);
  const int nsup = N / (16 * NT);
  const int wm = wid / nsup, wn = wid % nsup;
  const int m0 = wm * 16, n0 = wn * 16 * NT;
  if (m0 >= M) return;

  v8f acc[NT];
#pragma unroll
  for (int t = 0; t < NT; t++)
#pragma unroll
    for (int i = 0; i < 8; i++) acc[t][i] = 0.0f;

  const int m = m0 + (lane & 15);
  for (int kb = 0; kb < K; kb += 32) {
    v16bf af = load_a_frag(A, K, m, kb, lane);
#pragma unroll
    for (int t = 0; t < NT; t++) {
      const int n = n0 + t * 16 + (lane & 15);
      v16bf bf = load_b_frag(Bcol, K, n, kb, lane);
      acc[t] = wmma_bf16f32(af, bf, acc[t]);
    }
  }

  const int half = lane >> 4;
#pragma unroll
  for (int t = 0; t < NT; t++) {
    const int n = n0 + t * 16 + (lane & 15);
    const float bv = bias[n];
#pragma unroll
    for (int v = 0; v < 8; v++) {
      const int mm = m0 + half * 8 + v;
      float val = acc[t][v] + bv;
      if (RELU) val = fmaxf(val, 0.0f);
      if (OUTBF) outb[mm * N + n] = (__bf16)val;
      else       outf[mm * N + n] = val;
    }
  }
}

// ---------------------------------------------------------------------------
// B = [h[1:], h[:-1]] @ WB + bB  (M = T-1, K = 2*DH, N = DZ*DZ)
// ---------------------------------------------------------------------------
template <int NT>
__global__ __launch_bounds__(128) void wmma_gemm_hcat_kernel(
    const __bf16* __restrict__ h, const __bf16* __restrict__ WBcol,
    const float* __restrict__ bias, __bf16* __restrict__ outB) {
  const int N = DZ * DZ, K = 2 * DH, M = T_LEN - 1;
  const int lane = threadIdx.x & 31;
  const int wid  = blockIdx.x * (blockDim.x >> 5) + (threadIdx.x >> 5);
  const int nsup = N / (16 * NT);
  const int wm = wid / nsup, wn = wid % nsup;
  const int m0 = wm * 16, n0 = wn * 16 * NT;

  v8f acc[NT];
#pragma unroll
  for (int t = 0; t < NT; t++)
#pragma unroll
    for (int i = 0; i < 8; i++) acc[t][i] = 0.0f;

  const int trow = m0 + (lane & 15);
  const int r1 = (trow + 1 < T_LEN) ? trow + 1 : T_LEN - 1;  // clamp; masked store
  const int r0 = trow;

  for (int kb = 0; kb < DH; kb += 32) {           // h[t+1] x WB[0:DH]
    v16bf af = load_a_frag(h, DH, r1, kb, lane);
#pragma unroll
    for (int t = 0; t < NT; t++) {
      const int n = n0 + t * 16 + (lane & 15);
      v16bf bf = load_b_frag(WBcol, K, n, kb, lane);
      acc[t] = wmma_bf16f32(af, bf, acc[t]);
    }
  }
  for (int kb = 0; kb < DH; kb += 32) {           // h[t] x WB[DH:2DH]
    v16bf af = load_a_frag(h, DH, r0, kb, lane);
#pragma unroll
    for (int t = 0; t < NT; t++) {
      const int n = n0 + t * 16 + (lane & 15);
      v16bf bf = load_b_frag(WBcol, K, n, DH + kb, lane);
      acc[t] = wmma_bf16f32(af, bf, acc[t]);
    }
  }

  const int half = lane >> 4;
#pragma unroll
  for (int t = 0; t < NT; t++) {
    const int n = n0 + t * 16 + (lane & 15);
    const float bv = bias[n];
#pragma unroll
    for (int v = 0; v < 8; v++) {
      const int mm = m0 + half * 8 + v;
      if (mm < M) outB[mm * N + n] = (__bf16)(acc[t][v] + bv);
    }
  }
}

// ---------------------------------------------------------------------------
// Batched PSD products, one wave per t (K=32 == one WMMA per quadrant):
//   AA[t] = A A^T + (t>0 ? B_prev B_prev^T : 0) + I ;  BB[t] = A B^T
// ---------------------------------------------------------------------------
__global__ __launch_bounds__(128) void psd_kernel(
    const __bf16* __restrict__ Abf, const __bf16* __restrict__ Bbf,
    float* __restrict__ AAout, float* __restrict__ BBout) {
  const int lane = threadIdx.x & 31;
  const int t = blockIdx.x * (blockDim.x >> 5) + (threadIdx.x >> 5);
  if (t >= T_LEN) return;
  const __bf16* At = Abf + t * (DZ * DZ);
  const __bf16* Bp = Bbf + (t - 1) * (DZ * DZ);   // valid iff t > 0
  const int half = lane >> 4;

#pragma unroll
  for (int mi = 0; mi < 2; mi++) {
#pragma unroll
    for (int ni = 0; ni < 2; ni++) {
      const int m = mi * 16 + (lane & 15);
      const int n = ni * 16 + (lane & 15);
      v8f acc;
#pragma unroll
      for (int i = 0; i < 8; i++) acc[i] = 0.0f;
      acc = wmma_bf16f32(load_a_frag(At, DZ, m, 0, lane),
                         load_b_frag(At, DZ, n, 0, lane), acc);
      if (t > 0)
        acc = wmma_bf16f32(load_a_frag(Bp, DZ, m, 0, lane),
                           load_b_frag(Bp, DZ, n, 0, lane), acc);
#pragma unroll
      for (int v = 0; v < 8; v++) {
        const int mm = mi * 16 + half * 8 + v;
        AAout[t * (DZ * DZ) + mm * DZ + n] = acc[v] + ((mm == n) ? 1.0f : 0.0f);
      }
    }
  }

  if (t < T_LEN - 1) {
    const __bf16* Bt = Bbf + t * (DZ * DZ);
#pragma unroll
    for (int mi = 0; mi < 2; mi++) {
#pragma unroll
      for (int ni = 0; ni < 2; ni++) {
        const int m = mi * 16 + (lane & 15);
        const int n = ni * 16 + (lane & 15);
        v8f acc;
#pragma unroll
        for (int i = 0; i < 8; i++) acc[i] = 0.0f;
        acc = wmma_bf16f32(load_a_frag(At, DZ, m, 0, lane),
                           load_b_frag(Bt, DZ, n, 0, lane), acc);
#pragma unroll
        for (int v = 0; v < 8; v++) {
          const int mm = mi * 16 + half * 8 + v;
          BBout[t * (DZ * DZ) + mm * DZ + n] = acc[v];
        }
      }
    }
  }
}

// ---------------------------------------------------------------------------
// Sequential block-tridiagonal Cholesky (8191 dependent steps, one wave32).
// Lane = row, rows register-resident; column pivots broadcast via LDS.
// TDM double-buffer: while step t computes, TDM streams AA[t+1] and BB[t]
// into the other LDS buffer; s_wait_tensorcnt(2) = current pair landed.
// ---------------------------------------------------------------------------
__device__ __forceinline__ void chol_step(float (&Mr)[DZ], float (&Lr)[DZ],
                                          float* Lsh, int r) {
#pragma unroll
  for (int j = 0; j < DZ; j++) {
    float s = Mr[j];
#pragma unroll
    for (int k = 0; k < j; k++) s -= Lr[k] * Lsh[j * 33 + k];
    if (r == j) Lsh[j * 33 + j] = sqrtf(s);
    __syncthreads();
    const float d = Lsh[j * 33 + j];
    const float v = (r > j) ? (s / d) : ((r == j) ? d : 0.0f);
    Lr[j] = v;
    Lsh[r * 33 + j] = v;
    __syncthreads();
  }
}

__device__ __forceinline__ void load_row(const float* p, float (&row)[DZ]) {
#pragma unroll
  for (int q = 0; q < 8; q++) {
    float4 v = ((const float4*)p)[q];
    row[4 * q + 0] = v.x; row[4 * q + 1] = v.y;
    row[4 * q + 2] = v.z; row[4 * q + 3] = v.w;
  }
}
__device__ __forceinline__ void store_row(float* p, const float (&row)[DZ]) {
#pragma unroll
  for (int q = 0; q < 8; q++)
    ((float4*)p)[q] = make_float4(row[4 * q + 0], row[4 * q + 1],
                                  row[4 * q + 2], row[4 * q + 3]);
}

__global__ __launch_bounds__(32) void chol_kernel(float* __restrict__ cholA,
                                                  float* __restrict__ cholB) {
  __shared__ float AAbuf[2][DZ * DZ];
  __shared__ float BBbuf[2][DZ * DZ];
  __shared__ float Lsh[DZ * 33];
  __shared__ float Csh[DZ * 33];
  const int r = threadIdx.x;
  float Mrow[DZ], Lrow[DZ], Crow[DZ];

  // Prime TDM pipeline: AA[0]->buf0, AA[1]->buf1, BB[0]->buf1.
  tdm_load_block(cholA,        (unsigned)(uintptr_t)&AAbuf[0][0]);
  tdm_load_block(cholA + 1024, (unsigned)(uintptr_t)&AAbuf[1][0]);
  tdm_load_block(cholB,        (unsigned)(uintptr_t)&BBbuf[1][0]);
  __builtin_amdgcn_s_wait_tensorcnt(2);   // AA[0] landed (in-order)
  asm volatile("" ::: "memory");

  load_row(&AAbuf[0][r * DZ], Mrow);
  chol_step(Mrow, Lrow, Lsh, r);
  store_row(cholA + r * DZ, Lrow);

#pragma unroll 1
  for (int t = 1; t < T_LEN; t++) {
    const int cur = t & 1, nxt = cur ^ 1;
    if (t + 1 < T_LEN) {                  // stream next pair behind the math
      tdm_load_block(cholA + (t + 1) * (DZ * DZ), (unsigned)(uintptr_t)&AAbuf[nxt][0]);
      tdm_load_block(cholB + t * (DZ * DZ),       (unsigned)(uintptr_t)&BBbuf[nxt][0]);
      __builtin_amdgcn_s_wait_tensorcnt(2);  // all but newest 2 complete
    } else {
      __builtin_amdgcn_s_wait_tensorcnt(0);
    }
    asm volatile("" ::: "memory");

    // C = BB[t-1] * L^{-T} (forward solve per row; Lsh holds L_{t-1})
    float Brow[DZ];
    load_row(&BBbuf[cur][r * DZ], Brow);
#pragma unroll
    for (int j = 0; j < DZ; j++) {
      float s = Brow[j];
#pragma unroll
      for (int k = 0; k < j; k++) s -= Crow[k] * Lsh[j * 33 + k];
      Crow[j] = s / Lsh[j * 33 + j];
    }
    store_row(cholB + (t - 1) * (DZ * DZ) + r * DZ, Crow);
#pragma unroll
    for (int k = 0; k < DZ; k++) Csh[r * 33 + k] = Crow[k];
    __syncthreads();

    // M = AA[t] - C C^T
    load_row(&AAbuf[cur][r * DZ], Mrow);
#pragma unroll
    for (int c = 0; c < DZ; c++) {
      float s = Mrow[c];
#pragma unroll
      for (int k = 0; k < DZ; k++) s -= Crow[k] * Csh[c * 33 + k];
      Mrow[c] = s;
    }
    __syncthreads();

    chol_step(Mrow, Lrow, Lsh, r);
    store_row(cholA + t * (DZ * DZ) + r * DZ, Lrow);
  }
}

// ---------------------------------------------------------------------------
extern "C" void kernel_launch(void* const* d_in, const int* in_sizes, int n_in,
                              void* d_out, int out_size, void* d_ws, size_t ws_size,
                              hipStream_t stream) {
  (void)in_sizes; (void)n_in; (void)out_size; (void)ws_size;
  const float* X   = (const float*)d_in[0];
  const float* W1  = (const float*)d_in[1];
  const float* b1  = (const float*)d_in[2];
  const float* Wmu = (const float*)d_in[3];
  const float* bmu = (const float*)d_in[4];
  const float* WA  = (const float*)d_in[5];
  const float* bA  = (const float*)d_in[6];
  const float* WB  = (const float*)d_in[7];
  const float* bB  = (const float*)d_in[8];

  float* out    = (float*)d_out;
  float* mu_out = out;                               // (T, DZ)
  float* cholA  = out + T_LEN * DZ;                  // (T, DZ*DZ): AA then L
  float* cholB  = cholA + T_LEN * DZ * DZ;           // (T-1, DZ*DZ): BB then C

  char* ws = (char*)d_ws;
  __bf16* Xbf  = (__bf16*)ws; ws += (size_t)T_LEN * DH * 2;
  __bf16* W1t  = (__bf16*)ws; ws += (size_t)DH * DH * 2;
  __bf16* Wmut = (__bf16*)ws; ws += (size_t)DZ * DH * 2;
  __bf16* WAt  = (__bf16*)ws; ws += (size_t)(DZ * DZ) * DH * 2;
  __bf16* WBt  = (__bf16*)ws; ws += (size_t)(DZ * DZ) * 2 * DH * 2;
  __bf16* hbf  = (__bf16*)ws; ws += (size_t)T_LEN * DH * 2;
  __bf16* Abf  = (__bf16*)ws; ws += (size_t)T_LEN * DZ * DZ * 2;
  __bf16* Bbf  = (__bf16*)ws; ws += (size_t)T_LEN * DZ * DZ * 2;

  const int TPB = 256;
  auto eb = [](int n, int tpb) { return (n + tpb - 1) / tpb; };

  cvt_bf16_kernel<<<eb(T_LEN * DH, TPB), TPB, 0, stream>>>(X, Xbf, T_LEN * DH);
  cvt_transpose_kernel<<<eb(DH * DH, TPB), TPB, 0, stream>>>(W1, W1t, DH, DH);
  cvt_transpose_kernel<<<eb(DH * DZ, TPB), TPB, 0, stream>>>(Wmu, Wmut, DH, DZ);
  cvt_transpose_kernel<<<eb(DH * DZ * DZ, TPB), TPB, 0, stream>>>(WA, WAt, DH, DZ * DZ);
  cvt_transpose_kernel<<<eb(2 * DH * DZ * DZ, TPB), TPB, 0, stream>>>(WB, WBt, 2 * DH, DZ * DZ);

  auto wblocks = [](int waves) { return (waves + 3) / 4; };  // 128-thread blocks

  wmma_gemm_kernel<4, true, true><<<wblocks((T_LEN / 16) * (DH / 64)), 128, 0, stream>>>(
      Xbf, W1t, b1, nullptr, hbf, T_LEN, DH, DH);
  wmma_gemm_kernel<2, false, false><<<wblocks((T_LEN / 16) * (DZ / 32)), 128, 0, stream>>>(
      hbf, Wmut, bmu, mu_out, nullptr, T_LEN, DZ, DH);
  wmma_gemm_kernel<4, false, true><<<wblocks((T_LEN / 16) * (DZ * DZ / 64)), 128, 0, stream>>>(
      hbf, WAt, bA, nullptr, Abf, T_LEN, DZ * DZ, DH);
  wmma_gemm_hcat_kernel<4><<<wblocks((T_LEN / 16) * (DZ * DZ / 64)), 128, 0, stream>>>(
      hbf, WBt, bB, Bbf);
  psd_kernel<<<wblocks(T_LEN), 128, 0, stream>>>(Abf, Bbf, cholA, cholB);
  chol_kernel<<<1, 32, 0, stream>>>(cholA, cholB);
}
